// MoELayer_8907762172516
// MI455X (gfx1250) — compile-verified, matrix operand order checked by reading, not run on hardware
//
#include <hip/hip_runtime.h>
#include <hip/hip_bf16.h>

// ---------------- problem constants ----------------
#define BSZ   16384      // batch
#define DIN   512
#define NE    16         // experts
#define HID   512        // hidden
#define DT    4112       // total_input_size
#define DTP   4128       // DT padded to multiple of 32 (K for GEMM1)

// ---------------- GEMM tiling ----------------
#define BM    256
#define BN    128
#define BK    32         // LDS row pitch == BK (2-stage LDS = 48KB)

typedef __attribute__((ext_vector_type(16))) __bf16        v16bf;
typedef __attribute__((ext_vector_type(8)))  float         v8f;
typedef __attribute__((ext_vector_type(4)))  unsigned int  v4u;

union Frag { v16bf v; v4u q[2]; };

static __device__ __forceinline__ unsigned short f32_to_bf16(float f) {
    unsigned u = __float_as_uint(f);
    u += 0x7FFFu + ((u >> 16) & 1u);          // round-to-nearest-even
    return (unsigned short)(u >> 16);
}
static __device__ __forceinline__ float bf16_to_f32(unsigned short h) {
    return __uint_as_float(((unsigned)h) << 16);
}
// low 32 bits of a generic pointer to __shared__ data == LDS byte offset
static __device__ __forceinline__ unsigned lds_off(const void* p) {
    return (unsigned)(unsigned long long)p;
}

// CDNA5 async copy: 16B global -> LDS at vdst, same offset added to both addresses, tracked by ASYNCcnt
#define ASYNC_B128(ldsoff, gsrc, OFF)                                          \
    asm volatile("global_load_async_to_lds_b128 %0, %1, off offset:" #OFF      \
                 :: "v"(ldsoff), "v"(gsrc) : "memory")

static __device__ __forceinline__ void issue_tile(unsigned aDst, const unsigned short* aS,
                                                  unsigned bDst, const unsigned short* bS) {
    // A: 64B row per thread (4 x b128); B: 32B half-row per thread (2 x b128)
    ASYNC_B128(aDst, aS, 0);
    ASYNC_B128(aDst, aS, 16);
    ASYNC_B128(aDst, aS, 32);
    ASYNC_B128(aDst, aS, 48);
    ASYNC_B128(bDst, bS, 0);
    ASYNC_B128(bDst, bS, 16);
}

// one BK-step of WMMA math from an LDS stage buffer
static __device__ __forceinline__ void compute_step(const unsigned short* As,
                                                    const unsigned short* Bs,
                                                    int waveM, int waveN, int l, int halfB,
                                                    v8f acc[4][4]) {
    Frag a[4], bfr[4];
    const char* Ab = (const char*)As;
    const char* Bb = (const char*)Bs;
#pragma unroll
    for (int mi = 0; mi < 4; ++mi) {
        const int row = waveM * 64 + mi * 16 + l;
        a[mi].q[0] = *(const v4u*)(Ab + row * (BK * 2) + halfB);
        a[mi].q[1] = *(const v4u*)(Ab + row * (BK * 2) + 32 + halfB);
    }
#pragma unroll
    for (int ni = 0; ni < 4; ++ni) {
        const int col = waveN * 64 + ni * 16 + l;
        bfr[ni].q[0] = *(const v4u*)(Bb + col * (BK * 2) + halfB);
        bfr[ni].q[1] = *(const v4u*)(Bb + col * (BK * 2) + 32 + halfB);
    }
#pragma unroll
    for (int mi = 0; mi < 4; ++mi)
#pragma unroll
        for (int ni = 0; ni < 4; ++ni)
            acc[mi][ni] = __builtin_amdgcn_wmma_f32_16x16x32_bf16(
                false, a[mi].v, false, bfr[ni].v, (short)0, acc[mi][ni], false, false);
}

// ---------------- gate: [B,16] -> relu(16x8) -> (8x16) -> softmax ----------------
__global__ void gate_kernel(const float* __restrict__ cate,
                            const float* __restrict__ w1, const float* __restrict__ b1,
                            const float* __restrict__ w2, const float* __restrict__ b2,
                            float* __restrict__ gates) {
    int b = blockIdx.x * blockDim.x + threadIdx.x;
    if (b >= BSZ) return;
    float c[16];
#pragma unroll
    for (int i = 0; i < 16; ++i) c[i] = cate[b * 16 + i];
    float h[8];
#pragma unroll
    for (int j = 0; j < 8; ++j) {
        float s = b1[j];
#pragma unroll
        for (int i = 0; i < 16; ++i) s += c[i] * w1[i * 8 + j];
        h[j] = fmaxf(s, 0.f);
    }
    float lg[NE], mx = -3.4e38f;
#pragma unroll
    for (int e = 0; e < NE; ++e) {
        float s = b2[e];
#pragma unroll
        for (int j = 0; j < 8; ++j) s += h[j] * w2[j * NE + e];
        lg[e] = s; mx = fmaxf(mx, s);
    }
    float den = 0.f;
#pragma unroll
    for (int e = 0; e < NE; ++e) { lg[e] = __expf(lg[e] - mx); den += lg[e]; }
    float inv = 1.f / den;
#pragma unroll
    for (int e = 0; e < NE; ++e) gates[b * NE + e] = lg[e] * inv;
}

// ---------------- build combined bf16 [B, DTP] (zero-padded K) ----------------
__global__ void cvt_combined(const float* __restrict__ s0, const float* __restrict__ s1,
                             const float* __restrict__ s2, const float* __restrict__ s3,
                             const float* __restrict__ s4, const float* __restrict__ s5,
                             const float* __restrict__ s6, const float* __restrict__ s7,
                             const float* __restrict__ cate,
                             unsigned short* __restrict__ out) {
    size_t idx = (size_t)blockIdx.x * blockDim.x + threadIdx.x;
    size_t b = idx / DTP;
    int    d = (int)(idx % DTP);
    float v;
    if (d < 4096) {
        const float* srcs[8] = {s0, s1, s2, s3, s4, s5, s6, s7};
        v = srcs[d >> 9][b * DIN + (d & 511)];
    } else if (d < DT) {
        v = cate[b * 16 + (d - 4096)];
    } else {
        v = 0.f;
    }
    out[idx] = f32_to_bf16(v);
}

// ---------------- w1t[e][n][k] = bf16(ew1[e][k][n]), K padded ----------------
__global__ void cvt_w1t(const float* __restrict__ ew1, unsigned short* __restrict__ out) {
    size_t idx = (size_t)blockIdx.x * blockDim.x + threadIdx.x;
    size_t e = idx / ((size_t)HID * DTP);
    size_t r = idx % ((size_t)HID * DTP);
    size_t n = r / DTP;
    size_t k = r % DTP;
    float v = (k < DT) ? ew1[(e * DT + k) * HID + n] : 0.f;
    out[idx] = f32_to_bf16(v);
}

// ---------------- w2t[e][n][k] = bf16(ew2[e][k][n]) ----------------
__global__ void cvt_w2t(const float* __restrict__ ew2, unsigned short* __restrict__ out) {
    size_t idx = (size_t)blockIdx.x * blockDim.x + threadIdx.x;
    size_t e = idx / ((size_t)HID * HID);
    size_t r = idx % ((size_t)HID * HID);
    size_t n = r / HID;
    size_t k = r % HID;
    out[idx] = f32_to_bf16(ew2[(e * HID + k) * HID + n]);
}

// ---------------- WMMA GEMM: C[e] = relu(A[e] @ Bt[e]^T + bias[e]), bf16 out ----------------
// A: [M, K] bf16 row-major; Bt: [N, K] bf16 row-major (B transposed); C: [M, HID] bf16.
// Ping-pong LDS double buffering with compile-time buffer parity; tiles filled with
// global_load_async_to_lds_b128 (ASYNCcnt), 6 ops/stage/thread, in-order completion.
__global__ __launch_bounds__(256)
void wmma_gemm_bias_relu(const unsigned short* __restrict__ A,
                         const unsigned short* __restrict__ Bt,
                         const float* __restrict__ bias,
                         unsigned short* __restrict__ C,
                         int K, size_t strideA, size_t strideB, size_t strideC) {
    const int e = blockIdx.z;
    const unsigned short* Ae = A  + (size_t)e * strideA;
    const unsigned short* Be = Bt + (size_t)e * strideB;
    const float*          be = bias + (size_t)e * HID;
    unsigned short*       Ce = C  + (size_t)e * strideC;

    const int m0 = blockIdx.x * BM;
    const int n0 = blockIdx.y * BN;
    const int tid   = threadIdx.x;
    const int lane  = tid & 31;
    const int wave  = tid >> 5;
    const int waveM = wave >> 1;          // 0..3 -> 64-row slab
    const int waveN = wave & 1;           // 0..1 -> 64-col slab
    const int l     = lane & 15;
    const int halfB = (lane >> 4) * 16;   // byte offset: K 0..7 vs 8..15 halves

    __shared__ __align__(16) unsigned short As[2][BM * BK];  // 2 x 16KB
    __shared__ __align__(16) unsigned short Bs[2][BN * BK];  // 2 x  8KB

    // per-thread copy source pointers (advance by BK each K-step)
    const unsigned short* aSrc = Ae + (size_t)(m0 + tid) * K;
    const int br = tid >> 1, bh = tid & 1;
    const unsigned short* bSrc = Be + (size_t)(n0 + br) * K + bh * 16;

    const unsigned aDst0 = lds_off(&As[0][tid * BK]);
    const unsigned aDst1 = lds_off(&As[1][tid * BK]);
    const unsigned bDst0 = lds_off(&Bs[0][br * BK + bh * 16]);
    const unsigned bDst1 = lds_off(&Bs[1][br * BK + bh * 16]);

    v8f acc[4][4] = {};

    const int ksteps = K / BK;
    issue_tile(aDst0, aSrc, bDst0, bSrc);                 // stage 0 -> buf0

    int kt = 0;
    while (kt + 2 <= ksteps) {
        // ---- stage kt (buf0); prefetch kt+1 -> buf1 ----
        {
            const size_t off1 = (size_t)(kt + 1) * BK;
            issue_tile(aDst1, aSrc + off1, bDst1, bSrc + off1);
            asm volatile("s_wait_asynccnt 0x6" ::: "memory");   // oldest 6 (= stage kt) done
            __syncthreads();
            compute_step(As[0], Bs[0], waveM, waveN, l, halfB, acc);
            __syncthreads();                                    // buf0 free
        }
        // ---- stage kt+1 (buf1); prefetch kt+2 -> buf0 (clamped dummy past end) ----
        {
            const size_t off2 = (kt + 2 < ksteps) ? (size_t)(kt + 2) * BK : 0;
            issue_tile(aDst0, aSrc + off2, bDst0, bSrc + off2);
            asm volatile("s_wait_asynccnt 0x6" ::: "memory");   // stage kt+1 done
            __syncthreads();
            compute_step(As[1], Bs[1], waveM, waveN, l, halfB, acc);
            __syncthreads();                                    // buf1 free
        }
        kt += 2;
    }
    if (kt < ksteps) {                                          // odd tail: stage in buf0
        asm volatile("s_wait_asynccnt 0x0" ::: "memory");
        __syncthreads();
        compute_step(As[0], Bs[0], waveM, waveN, l, halfB, acc);
    }
    // any dummy prefetch drains at s_endpgm (implicit wait-idle)

    // epilogue: bias + relu + bf16 store. C frag: VGPR r -> row r (+8 for hi half), lane&15 -> col.
    const int rbase = (lane >> 4) * 8;
#pragma unroll
    for (int mi = 0; mi < 4; ++mi) {
#pragma unroll
        for (int ni = 0; ni < 4; ++ni) {
            const int gcol  = n0 + waveN * 64 + ni * 16 + l;
            const float bv  = be[gcol];
            const int grow0 = m0 + waveM * 64 + mi * 16 + rbase;
#pragma unroll
            for (int r = 0; r < 8; ++r) {
                float v = acc[mi][ni][r] + bv;
                v = fmaxf(v, 0.f);
                Ce[(size_t)(grow0 + r) * HID + gcol] = f32_to_bf16(v);
            }
        }
    }
}

// ---------------- final: out[b] = sum_e gates[b,e] * (h2[e,b,:] . ew3[e,:] + eb3[e]) ----------------
__global__ __launch_bounds__(256)
void out_kernel(const unsigned short* __restrict__ h2, const float* __restrict__ gates,
                const float* __restrict__ ew3, const float* __restrict__ eb3,
                float* __restrict__ out) {
    const int gtid = blockIdx.x * blockDim.x + threadIdx.x;
    const int b    = gtid >> 5;           // one wave32 per row
    const int lane = gtid & 31;
    if (b >= BSZ) return;
    float total = 0.f;
    for (int e = 0; e < NE; ++e) {
        const unsigned short* hp = h2 + ((size_t)e * BSZ + b) * HID;
        const float*          wp = ew3 + (size_t)e * HID;
        float s = 0.f;
        for (int h = lane; h < HID; h += 32) s += bf16_to_f32(hp[h]) * wp[h];
#pragma unroll
        for (int off = 16; off > 0; off >>= 1) s += __shfl_xor(s, off, 32);
        total += gates[b * NE + e] * (s + eb3[e]);
    }
    if (lane == 0) out[b] = total;
}

extern "C" void kernel_launch(void* const* d_in, const int* in_sizes, int n_in,
                              void* d_out, int out_size, void* d_ws, size_t ws_size,
                              hipStream_t stream) {
    const float* user_emb = (const float*)d_in[0];
    const float* item_emb = (const float*)d_in[1];
    const float* f_i_1    = (const float*)d_in[2];
    const float* g_i_1    = (const float*)d_in[3];
    const float* f_i_2    = (const float*)d_in[4];
    const float* g_i_2    = (const float*)d_in[5];
    const float* f_i_3    = (const float*)d_in[6];
    const float* g_i_3    = (const float*)d_in[7];
    const float* cate_emb = (const float*)d_in[8];
    const float* gate_w1  = (const float*)d_in[9];
    const float* gate_b1  = (const float*)d_in[10];
    const float* gate_w2  = (const float*)d_in[11];
    const float* gate_b2  = (const float*)d_in[12];
    const float* ew1      = (const float*)d_in[13];
    const float* eb1      = (const float*)d_in[14];
    const float* ew2      = (const float*)d_in[15];
    const float* eb2      = (const float*)d_in[16];
    const float* ew3      = (const float*)d_in[17];
    const float* eb3      = (const float*)d_in[18];
    float* out = (float*)d_out;

    // workspace layout (all sizes multiples of 256 bytes)
    char* base = (char*)d_ws;
    size_t off = 0;
    float* gates = (float*)(base + off);                      off += (size_t)BSZ * NE * sizeof(float);
    unsigned short* combined = (unsigned short*)(base + off); off += (size_t)BSZ * DTP * 2;
    unsigned short* w1t      = (unsigned short*)(base + off); off += (size_t)NE * HID * DTP * 2;
    unsigned short* w2t      = (unsigned short*)(base + off); off += (size_t)NE * HID * HID * 2;
    unsigned short* h1       = (unsigned short*)(base + off); off += (size_t)NE * BSZ * HID * 2;
    unsigned short* h2       = (unsigned short*)(base + off); off += (size_t)NE * BSZ * HID * 2;
    (void)ws_size; (void)in_sizes; (void)n_in; (void)out_size;

    // 1) gate
    gate_kernel<<<BSZ / 256, 256, 0, stream>>>(cate_emb, gate_w1, gate_b1, gate_w2, gate_b2, gates);
    // 2) downconvert activations + weights (bf16, weights pre-transposed to [n][k])
    cvt_combined<<<(int)(((size_t)BSZ * DTP) / 256), 256, 0, stream>>>(
        user_emb, item_emb, f_i_1, g_i_1, f_i_2, g_i_2, f_i_3, g_i_3, cate_emb, combined);
    cvt_w1t<<<(int)(((size_t)NE * HID * DTP) / 256), 256, 0, stream>>>(ew1, w1t);
    cvt_w2t<<<(int)(((size_t)NE * HID * HID) / 256), 256, 0, stream>>>(ew2, w2t);

    // 3) layer 1: h1[e] = relu(combined @ ew1[e] + eb1[e]), K = DTP
    dim3 grid1(BSZ / BM, HID / BN, NE);
    wmma_gemm_bias_relu<<<grid1, 256, 0, stream>>>(
        combined, w1t, eb1, h1, DTP,
        /*strideA=*/0, /*strideB=*/(size_t)HID * DTP, /*strideC=*/(size_t)BSZ * HID);

    // 4) layer 2: h2[e] = relu(h1[e] @ ew2[e] + eb2[e]), K = HID
    wmma_gemm_bias_relu<<<grid1, 256, 0, stream>>>(
        h1, w2t, eb2, h2, HID,
        /*strideA=*/(size_t)BSZ * HID, /*strideB=*/(size_t)HID * HID, /*strideC=*/(size_t)BSZ * HID);

    // 5) layer 3 + gate-weighted combine
    out_kernel<<<(BSZ * 32) / 256, 256, 0, stream>>>(h2, gates, ew3, eb3, out);
}
